// SelfAttention2d_16887811408533
// MI455X (gfx1250) — matrix operand order repaired and checked by reading, not compiled, not run on hardware
//
#include <hip/hip_runtime.h>
#include <hip/hip_bf16.h>

// ---------------------------------------------------------------------------
// Self-attention 2d block for MI455X (gfx1250, wave32, WMMA bf16)
// B=256, C=512, H=W=14 (N=196, padded to 224), heads=8, head_dim=64, groups=32
// ---------------------------------------------------------------------------

typedef __attribute__((ext_vector_type(8)))  __bf16 v8bf;
typedef __attribute__((ext_vector_type(16))) __bf16 v16bf;
typedef __attribute__((ext_vector_type(8)))  float  v8f;
typedef __attribute__((ext_vector_type(4)))  int    v4i;

#define BB 256
#define CC 512
#define NN 196
#define NP 224           // padded spatial (divisible by 32)
#define NHEADS 8
#define HDIM 64
#define NGROUPS 32
#define GSIZE 16         // channels per group
#define OC_QKV 1536
#define SCALE 0.125f     // 64^-0.5

// gfx1250 async global->LDS path (guarded; falls back to plain LDS copy)
#if defined(__has_builtin)
#  if __has_builtin(__builtin_amdgcn_global_load_async_to_lds_b128) && \
      __has_builtin(__builtin_amdgcn_s_wait_asynccnt)
#    define USE_ASYNC_LDS 1
#  endif
#endif

__device__ __forceinline__ v16bf bfcat(v8bf lo, v8bf hi) {
  return __builtin_shufflevector(lo, hi, 0,1,2,3,4,5,6,7,8,9,10,11,12,13,14,15);
}
__device__ __forceinline__ v8f wmma_bf16(v16bf a, v16bf b, v8f c) {
  // D = A(16x32) * B(32x16) + C, fp32 accumulate
  return __builtin_amdgcn_wmma_f32_16x16x32_bf16(false, a, false, b, (short)0, c, false, false);
}
__device__ __forceinline__ v8bf ld8(const __bf16* p) { return *(const v8bf*)p; }

// stage one 16-byte chunk global -> LDS (per-thread)
__device__ __forceinline__ void stage16(const __bf16* src, __bf16* dst) {
#if defined(USE_ASYNC_LDS)
  __builtin_amdgcn_global_load_async_to_lds_b128(
      (__attribute__((address_space(1))) v4i*)(src),
      (__attribute__((address_space(3))) v4i*)(dst), 0, 0);
#else
  *(v4i*)dst = *(const v4i*)src;
#endif
}
__device__ __forceinline__ void stage_wait() {
#if defined(USE_ASYNC_LDS)
  __builtin_amdgcn_s_wait_asynccnt(0);
#endif
}

// ---------------------------------------------------------------------------
// Kernel 1: fp32 -> bf16 weight conversion (qkv_w: 1536x512, proj_w: 512x512)
// ---------------------------------------------------------------------------
__global__ void k_wconv(const float* __restrict__ qkvw, const float* __restrict__ projw,
                        __bf16* __restrict__ wq, __bf16* __restrict__ wp) {
  int i = blockIdx.x * 256 + threadIdx.x;             // 1,048,576 threads exactly
  if (i < OC_QKV * CC) wq[i] = (__bf16)qkvw[i];
  else                 wp[i - OC_QKV * CC] = (__bf16)projw[i - OC_QKV * CC];
}

// ---------------------------------------------------------------------------
// Kernel 2: GroupNorm statistics. One block per (b, group); reduce 16*196 vals.
// ---------------------------------------------------------------------------
__global__ void k_gnstats(const float* __restrict__ x, float* __restrict__ stats) {
  int b = blockIdx.x / NGROUPS, g = blockIdx.x % NGROUPS;
  const float* xp = x + ((size_t)(b * CC + g * GSIZE)) * NN;   // contiguous 3136 floats
  float s = 0.f, ss = 0.f;
  for (int i = threadIdx.x; i < GSIZE * NN; i += 256) {
    float v = xp[i];
    s += v; ss += v * v;
  }
  __shared__ float r1[256], r2[256];
  r1[threadIdx.x] = s; r2[threadIdx.x] = ss;
  __syncthreads();
  for (int off = 128; off > 0; off >>= 1) {
    if (threadIdx.x < off) { r1[threadIdx.x] += r1[threadIdx.x + off]; r2[threadIdx.x] += r2[threadIdx.x + off]; }
    __syncthreads();
  }
  if (threadIdx.x == 0) {
    const float invn = 1.f / (GSIZE * NN);
    float mean = r1[0] * invn;
    float var  = r2[0] * invn - mean * mean;
    stats[(b * NGROUPS + g) * 2 + 0] = mean;
    stats[(b * NGROUPS + g) * 2 + 1] = rsqrtf(var + 1e-5f);
  }
}

// ---------------------------------------------------------------------------
// Kernel 3: apply GroupNorm, pack bf16 into H[b][n_pad][c] (c contiguous = K)
// ---------------------------------------------------------------------------
__global__ void k_gnapply(const float* __restrict__ x, const float* __restrict__ stats,
                          const float* __restrict__ gw, const float* __restrict__ gb,
                          __bf16* __restrict__ H) {
  int idx = blockIdx.x * 256 + threadIdx.x;           // B*C*NP threads exactly
  int n = idx % NP;
  int c = (idx / NP) % CC;
  int b = idx / (NP * CC);
  float v = 0.f;
  if (n < NN) {
    int g = c >> 4;
    float mean = stats[(b * NGROUPS + g) * 2 + 0];
    float rstd = stats[(b * NGROUPS + g) * 2 + 1];
    v = (x[((size_t)(b * CC + c)) * NN + n] - mean) * rstd * gw[c] + gb[c];
  }
  H[((size_t)(b * NP + n)) * CC + c] = (__bf16)v;
}

// ---------------------------------------------------------------------------
// Block-cooperative GEMM main loop (shared by kernels 4 and 6):
//   block tile 64(M) x 256(N); A panel (64x32 bf16 per K-step) staged in LDS
//   via async global->LDS; 8 waves as 2(M) x 4(N), each wave 32x64 = 8 accs.
//   Defines: acc[2][4], lm, hi, om (wave M origin within block), colw.
// ---------------------------------------------------------------------------
#define GEMM_MAINLOOP(Wptr, Bsrc)                                               \
  __shared__ __align__(16) __bf16 ldsA[64 * 32];                                \
  v8f acc[2][4] = {{v8f{},v8f{},v8f{},v8f{}},{v8f{},v8f{},v8f{},v8f{}}};        \
  const int tid = threadIdx.x;                                                  \
  const int lane = tid & 31;                                                    \
  const int lm = lane & 15;                                                     \
  const int hi = (lane >> 4) & 1;                                               \
  const int wm = (tid >> 7) & 1;            /* wave M index: 0..1  */           \
  const int wn = (tid >> 5) & 3;            /* wave N index: 0..3  */           \
  const int om = wm * 32;                   /* wave M origin in block tile */   \
  const int colw = col0 + wn * 64;          /* wave N origin (global col) */    \
  const int selA = hi * 8, selB = hi * 8;                                       \
  const int arow = tid >> 2, aseg = (tid & 3) * 8;                              \
  for (int k0 = 0; k0 < CC; k0 += 32) {                                         \
    stage16(Wptr + (size_t)(o0 + arow) * CC + k0 + aseg,                        \
            &ldsA[arow * 32 + aseg]);                                           \
    stage_wait();                                                               \
    __syncthreads();                                                            \
    v16bf Aop[2];                                                               \
    _Pragma("unroll")                                                           \
    for (int os = 0; os < 2; ++os) {                                            \
      const __bf16* ar = &ldsA[(om + os * 16 + lm) * 32];                       \
      Aop[os] = bfcat(ld8(ar + selA), ld8(ar + 16 + selA));                     \
    }                                                                           \
    _Pragma("unroll")                                                           \
    for (int cs = 0; cs < 4; ++cs) {                                            \
      const __bf16* hcol = Bsrc + (size_t)(colw + cs * 16 + lm) * CC + k0 + selB;\
      v16bf Bv = bfcat(ld8(hcol), ld8(hcol + 16));                              \
      acc[0][cs] = wmma_bf16(Aop[0], Bv, acc[0][cs]);                           \
      acc[1][cs] = wmma_bf16(Aop[1], Bv, acc[1][cs]);                           \
    }                                                                           \
    __syncthreads();                                                            \
  }

// ---------------------------------------------------------------------------
// Kernel 4: QKV GEMM.  out[o, col] = sum_c W[o,c]*H[col,c] + bias[o]
//   grid: (1536/64) * (14336/256) = 24*56 = 1344 blocks of 256 threads.
//   Scatter-store into q[bh][n][d], k[bh][n][d], v[bh][d][n] (bf16).
// ---------------------------------------------------------------------------
__global__ void k_qkvgemm(const __bf16* __restrict__ W, const __bf16* __restrict__ H,
                          const float* __restrict__ bias,
                          __bf16* __restrict__ Q, __bf16* __restrict__ K, __bf16* __restrict__ V) {
  const int OBLK = OC_QKV / 64;               // 24
  int ob = blockIdx.x % OBLK;
  int cb = blockIdx.x / OBLK;                 // 0..55
  int o0 = ob * 64;
  int col0 = cb * 256;

  GEMM_MAINLOOP(W, H)

#pragma unroll
  for (int os = 0; os < 2; ++os) {
#pragma unroll
    for (int cs = 0; cs < 4; ++cs) {
      int col = colw + cs * 16 + lm;
      int b = col / NP, n = col % NP;
#pragma unroll
      for (int r = 0; r < 8; ++r) {
        int oe = o0 + om + os * 16 + r + hi * 8;
        float val = acc[os][cs][r] + bias[oe];
        int sec = oe >> 9;                    // 0=q 1=k 2=v
        int oc = oe & 511;
        int head = oc >> 6, d = oc & 63;
        int bh = b * NHEADS + head;
        __bf16 bv = (__bf16)val;
        if (sec == 0)      Q[((size_t)(bh * NP + n)) * HDIM + d] = bv;
        else if (sec == 1) K[((size_t)(bh * NP + n)) * HDIM + d] = bv;
        else               V[((size_t)(bh * HDIM + d)) * NP + n] = bv;
      }
    }
  }
}

// ---------------------------------------------------------------------------
// Kernel 5: attention, flash style. One wave (32 thr) per (b, head, n-tile).
// ---------------------------------------------------------------------------
__global__ void k_attn(const __bf16* __restrict__ Q, const __bf16* __restrict__ K,
                       const __bf16* __restrict__ V, __bf16* __restrict__ AO) {
  const int NT = NP / 16;                     // 14 n-tiles
  int blk = blockIdx.x;
  int nt = blk % NT;
  int head = (blk / NT) % NHEADS;
  int b = blk / (NT * NHEADS);
  int lane = threadIdx.x;
  int lm = lane & 15;
  int hi = (lane >> 4) & 1;
  int bh = b * NHEADS + head;

  const __bf16* qb = Q + (size_t)bh * NP * HDIM;
  const __bf16* kb = K + (size_t)bh * NP * HDIM;
  const __bf16* vb = V + (size_t)bh * HDIM * NP;

  const __bf16* qrow = qb + (size_t)(nt * 16 + lm) * HDIM;
  int selA = hi * 8;
  v16bf Aq0 = bfcat(ld8(qrow +  0 + selA), ld8(qrow + 16 + selA));
  v16bf Aq1 = bfcat(ld8(qrow + 32 + selA), ld8(qrow + 48 + selA));

  v8f o0 = {}, o1 = {}, o2 = {}, o3 = {};
  float rmax = -__builtin_inff(), rsum = 0.f;   // valid in lanes 0..15 (row = lane)

  __shared__ __align__(16) float  s_s[16 * 32];
  __shared__ __align__(16) __bf16 s_p[16 * 32];
  __shared__ float s_a[16];

  for (int mc = 0; mc < NP; mc += 32) {
    int selB = hi * 8;
    const __bf16* kr0 = kb + (size_t)(mc + lm) * HDIM;
    const __bf16* kr1 = kb + (size_t)(mc + 16 + lm) * HDIM;
    v8f s0 = {}, s1 = {};
    s0 = wmma_bf16(Aq0, bfcat(ld8(kr0 + selB),      ld8(kr0 + selB + 16)),      s0);
    s0 = wmma_bf16(Aq1, bfcat(ld8(kr0 + 32 + selB), ld8(kr0 + 32 + selB + 16)), s0);
    s1 = wmma_bf16(Aq0, bfcat(ld8(kr1 + selB),      ld8(kr1 + selB + 16)),      s1);
    s1 = wmma_bf16(Aq1, bfcat(ld8(kr1 + 32 + selB), ld8(kr1 + 32 + selB + 16)), s1);

#pragma unroll
    for (int r = 0; r < 8; ++r) {
      int nl = r + hi * 8;
      float v0 = (mc + lm      < NN) ? s0[r] * SCALE : -1e30f;
      float v1 = (mc + 16 + lm < NN) ? s1[r] * SCALE : -1e30f;
      s_s[nl * 32 + lm]      = v0;
      s_s[nl * 32 + 16 + lm] = v1;
    }
    __syncthreads();

    if (lane < 16) {                          // online softmax, lane = row
      float mx = rmax;
      for (int t = 0; t < 32; ++t) mx = fmaxf(mx, s_s[lane * 32 + t]);
      float alpha = __expf(rmax - mx);
      float sum = 0.f;
      for (int t = 0; t < 32; ++t) {
        float p = __expf(s_s[lane * 32 + t] - mx);
        sum += p;
        s_p[lane * 32 + t] = (__bf16)p;
      }
      rsum = rsum * alpha + sum;
      rmax = mx;
      s_a[lane] = alpha;
    }
    __syncthreads();

    float af[8];
#pragma unroll
    for (int r = 0; r < 8; ++r) af[r] = s_a[r + hi * 8];
#pragma unroll
    for (int r = 0; r < 8; ++r) { o0[r] *= af[r]; o1[r] *= af[r]; o2[r] *= af[r]; o3[r] *= af[r]; }

    const __bf16* prow = &s_p[lm * 32];
    v16bf Ap = bfcat(ld8(prow + selA), ld8(prow + 16 + selA));
    { const __bf16* vr = vb + (size_t)(0 * 16 + lm) * NP + mc + selB;
      o0 = wmma_bf16(Ap, bfcat(ld8(vr), ld8(vr + 16)), o0); }
    { const __bf16* vr = vb + (size_t)(1 * 16 + lm) * NP + mc + selB;
      o1 = wmma_bf16(Ap, bfcat(ld8(vr), ld8(vr + 16)), o1); }
    { const __bf16* vr = vb + (size_t)(2 * 16 + lm) * NP + mc + selB;
      o2 = wmma_bf16(Ap, bfcat(ld8(vr), ld8(vr + 16)), o2); }
    { const __bf16* vr = vb + (size_t)(3 * 16 + lm) * NP + mc + selB;
      o3 = wmma_bf16(Ap, bfcat(ld8(vr), ld8(vr + 16)), o3); }
    __syncthreads();
  }

  if (lane < 16) s_a[lane] = 1.f / rsum;
  __syncthreads();
  float fi[8];
#pragma unroll
  for (int r = 0; r < 8; ++r) fi[r] = s_a[r + hi * 8];
#pragma unroll
  for (int r = 0; r < 8; ++r) {
    int n = nt * 16 + r + hi * 8;
    size_t base = ((size_t)(b * NP + n)) * CC + head * HDIM + lm;
    AO[base + 0 * 16] = (__bf16)(o0[r] * fi[r]);
    AO[base + 1 * 16] = (__bf16)(o1[r] * fi[r]);
    AO[base + 2 * 16] = (__bf16)(o2[r] * fi[r]);
    AO[base + 3 * 16] = (__bf16)(o3[r] * fi[r]);
  }
}

// ---------------------------------------------------------------------------
// Kernel 6: proj GEMM + bias + residual -> d_out (fp32, [b][c][n], n<196)
//   grid: (512/64) * 56 = 448 blocks of 256 threads.
// ---------------------------------------------------------------------------
__global__ void k_projgemm(const __bf16* __restrict__ W, const __bf16* __restrict__ AO,
                           const float* __restrict__ bias, const float* __restrict__ x,
                           float* __restrict__ out) {
  const int OBLK = CC / 64;                   // 8
  int ob = blockIdx.x % OBLK;
  int cb = blockIdx.x / OBLK;                 // 0..55
  int o0 = ob * 64;
  int col0 = cb * 256;

  GEMM_MAINLOOP(W, AO)

#pragma unroll
  for (int os = 0; os < 2; ++os) {
#pragma unroll
    for (int cs = 0; cs < 4; ++cs) {
      int col = colw + cs * 16 + lm;
      int b = col / NP, n = col % NP;
      if (n < NN) {
#pragma unroll
        for (int r = 0; r < 8; ++r) {
          int oe = o0 + om + os * 16 + r + hi * 8;
          size_t oi = ((size_t)(b * CC + oe)) * NN + n;
          out[oi] = x[oi] + bias[oe] + acc[os][cs][r];
        }
      }
    }
  }
}

// ---------------------------------------------------------------------------
// Launch
// ---------------------------------------------------------------------------
extern "C" void kernel_launch(void* const* d_in, const int* in_sizes, int n_in,
                              void* d_out, int out_size, void* d_ws, size_t ws_size,
                              hipStream_t stream) {
  const float* x      = (const float*)d_in[0];
  const float* gn_w   = (const float*)d_in[1];
  const float* gn_b   = (const float*)d_in[2];
  const float* qkv_w  = (const float*)d_in[3];
  const float* qkv_b  = (const float*)d_in[4];
  const float* proj_w = (const float*)d_in[5];
  const float* proj_b = (const float*)d_in[6];
  float* out = (float*)d_out;

  const size_t SZ_STATS = (size_t)BB * NGROUPS * 2 * sizeof(float);      // 64 KB
  const size_t SZ_H     = (size_t)BB * NP * CC * sizeof(__bf16);         // 58.7 MB (reused as AO)
  const size_t SZ_QKV   = (size_t)BB * NHEADS * NP * HDIM * sizeof(__bf16);
  const size_t SZ_WQ    = (size_t)OC_QKV * CC * sizeof(__bf16);
  const size_t SZ_WP    = (size_t)CC * CC * sizeof(__bf16);

  char* p = (char*)d_ws;
  float*  stats = (float*)p;            p += SZ_STATS;
  __bf16* Hbuf  = (__bf16*)p;           p += SZ_H;      // h, later attn-out
  __bf16* Qb    = (__bf16*)p;           p += SZ_QKV;
  __bf16* Kb    = (__bf16*)p;           p += SZ_QKV;
  __bf16* Vb    = (__bf16*)p;           p += SZ_QKV;
  __bf16* Wq    = (__bf16*)p;           p += SZ_WQ;
  __bf16* Wp    = (__bf16*)p;           p += SZ_WP;
  if ((size_t)(p - (char*)d_ws) > ws_size) return;      // workspace too small

  k_wconv<<<4096, 256, 0, stream>>>(qkv_w, proj_w, Wq, Wp);
  k_gnstats<<<BB * NGROUPS, 256, 0, stream>>>(x, stats);
  k_gnapply<<<(BB * CC * NP) / 256, 256, 0, stream>>>(x, stats, gn_w, gn_b, Hbuf);
  // QKV GEMM: 24 * 56 = 1344 blocks
  k_qkvgemm<<<1344, 256, 0, stream>>>(Wq, Hbuf, qkv_b, Qb, Kb, Vb);
  // attention: 256*8*14 waves, 1 wave per block
  k_attn<<<BB * NHEADS * (NP / 16), 32, 0, stream>>>(Qb, Kb, Vb, Hbuf);
  // proj GEMM + residual: 8 * 56 = 448 blocks
  k_projgemm<<<448, 256, 0, stream>>>(Wp, Hbuf, proj_b, x, out);
}
